// Graph2VecSet2Set_54889682043381
// MI455X (gfx1250) — compile-verified, matrix operand order checked by reading, not run on hardware
//
#include <hip/hip_runtime.h>

#define N_NODES 50000
#define N_EDGES 625000
#define N_GRAPHS 256
#define D 128
#define STEPS 3

typedef __attribute__((ext_vector_type(16))) _Float16 v16h;
typedef __attribute__((ext_vector_type(8)))  float    v8f;

// ---------------- helpers ----------------
__device__ __forceinline__ float sigmoidf_(float x) { return 1.0f / (1.0f + __expf(-x)); }

// monotone float -> uint mapping (order-preserving) for atomicMax on floats
__device__ __forceinline__ unsigned fmap_(float f) {
    unsigned u = __float_as_uint(f);
    return (u & 0x80000000u) ? ~u : (u | 0x80000000u);
}
__device__ __forceinline__ float funmap_(unsigned u) {
    u = (u & 0x80000000u) ? (u & 0x7FFFFFFFu) : ~u;
    return __uint_as_float(u);
}

__global__ void fill_f32(float* __restrict__ p, float v, int n) {
    int i = blockIdx.x * blockDim.x + threadIdx.x;
    if (i < n) p[i] = v;
}

// ---------------- degree / normalization ----------------
__global__ void deg_kernel(const long long* __restrict__ ei, float* __restrict__ deg) {
    int e = blockIdx.x * blockDim.x + threadIdx.x;
    if (e >= N_EDGES) return;
    long long dst = ei[N_EDGES + e];
    atomicAdd(&deg[dst], 1.0f);
}

__global__ void rsqrt_kernel(float* __restrict__ d) {
    int i = blockIdx.x * blockDim.x + threadIdx.x;
    if (i < N_NODES) d[i] = rsqrtf(d[i]);
}

// ---------------- GEMM: Out[N,128] = X[N,128] @ W[128,128]  (f16 WMMA, f32 acc) ----------------
__global__ __launch_bounds__(256)
void gemm_wmma_kernel(const float* __restrict__ X, const float* __restrict__ W,
                      float* __restrict__ Out) {
    __shared__ _Float16 sA[16 * 132];    // 16 rows of X, padded
    __shared__ _Float16 sWt[128 * 136];  // W transposed: sWt[n][k], padded stride
    const int tid  = threadIdx.x;
    const int row0 = blockIdx.x * 16;    // 3125 * 16 == 50000 exactly

    for (int idx = tid; idx < D * D; idx += 256) {
        int k = idx >> 7, n = idx & 127;
        sWt[n * 136 + k] = (_Float16)W[idx];
    }
    for (int idx = tid; idx < 16 * D; idx += 256) {
        int r = idx >> 7, c = idx & 127;
        sA[r * 132 + c] = (_Float16)X[(size_t)(row0 + r) * D + c];
    }
    __syncthreads();

    const int wave  = tid >> 5;                 // 8 waves -> 8 column tiles of 16
    const int lane  = tid & 31;
    const int mrow  = lane & 15;
    const int khalf = (lane & 16) ? 8 : 0;      // 16-bit A-matrix lane layout
    const int ncol  = wave * 16 + (lane & 15);

    v8f acc = {};
    #pragma unroll
    for (int kt = 0; kt < 4; ++kt) {            // K = 128 in chunks of 32
        const int kb = kt * 32 + khalf;
        v16h a, b;
        #pragma unroll
        for (int j = 0; j < 8; ++j) {
            a[j]     = sA[mrow * 132 + kb + j];
            a[j + 8] = sA[mrow * 132 + kb + 16 + j];
            b[j]     = sWt[ncol * 136 + kb + j];
            b[j + 8] = sWt[ncol * 136 + kb + 16 + j];
        }
        acc = __builtin_amdgcn_wmma_f32_16x16x32_f16(
            /*neg_a=*/false, a, /*neg_b=*/false, b,
            /*c_mod=*/(short)0, acc, /*reuse_a=*/false, /*reuse_b=*/false);
    }

    const int rbase = (lane & 16) ? 8 : 0;      // C/D layout: lanes 16-31 hold M=8..15
    #pragma unroll
    for (int r = 0; r < 8; ++r) {
        Out[(size_t)(row0 + rbase + r) * D + ncol] = acc[r];
    }
}

// ---------------- edge scatter: agg[dst] += xw[src] * dinv[src]*dinv[dst] ----------------
__global__ void scatter_kernel(const float* __restrict__ xw, const long long* __restrict__ ei,
                               const float* __restrict__ dinv, float* __restrict__ agg) {
    int t = blockIdx.x * blockDim.x + threadIdx.x;
    int e = t >> 5;
    if (e >= N_EDGES) return;
    int lane = t & 31;

    long long s = ei[e];
    long long d = ei[N_EDGES + e];
    // prefetch a future source row into cache (emits global_prefetch_b8)
    if (e + 1024 < N_EDGES)
        __builtin_prefetch(xw + (size_t)ei[e + 1024] * D, 0, 1);

    float nrm = dinv[s] * dinv[d];
    const float4* xs = (const float4*)(xw + (size_t)s * D);
    float4 v = xs[lane];                         // 32 lanes x float4 = 128 cols
    float* ad = agg + (size_t)d * D + lane * 4;
    atomicAdd(ad + 0, v.x * nrm);
    atomicAdd(ad + 1, v.y * nrm);
    atomicAdd(ad + 2, v.z * nrm);
    atomicAdd(ad + 3, v.w * nrm);
}

// ---------------- self loop + bias (+relu) ----------------
__global__ void self_bias_kernel(float* __restrict__ agg, const float* __restrict__ xw,
                                 const float* __restrict__ dinv, const float* __restrict__ bias,
                                 int relu) {
    int i = blockIdx.x * blockDim.x + threadIdx.x;
    if (i >= N_NODES * D) return;
    int n = i >> 7, c = i & 127;
    float di = dinv[n];
    float v = agg[i] + xw[i] * di * di + bias[c];
    if (relu) v = fmaxf(v, 0.0f);
    agg[i] = v;
}

// ---------------- Set2Set ----------------
__global__ void gates_kernel(const float* __restrict__ qstar, const float* __restrict__ h,
                             const float* __restrict__ Wih, const float* __restrict__ Whh,
                             const float* __restrict__ bih, const float* __restrict__ bhh,
                             float* __restrict__ gates) {
    int t = blockIdx.x * blockDim.x + threadIdx.x;
    if (t >= N_GRAPHS * 4 * D) return;
    int b = t >> 9, j = t & 511;
    float acc = bih[j] + bhh[j];
    const float* q  = qstar + (size_t)b * 2 * D;
    const float* wi = Wih + (size_t)j * 2 * D;
    for (int k = 0; k < 2 * D; ++k) acc += q[k] * wi[k];
    const float* hb = h + (size_t)b * D;
    const float* wh = Whh + (size_t)j * D;
    for (int k = 0; k < D; ++k) acc += hb[k] * wh[k];
    gates[t] = acc;
}

__global__ void lstm_kernel(const float* __restrict__ gates, float* __restrict__ h,
                            float* __restrict__ c) {
    int t = blockIdx.x * blockDim.x + threadIdx.x;
    if (t >= N_GRAPHS * D) return;
    int b = t >> 7, d = t & 127;
    const float* g = gates + (size_t)b * 4 * D;
    float i_ = sigmoidf_(g[d]);
    float f_ = sigmoidf_(g[D + d]);
    float g_ = tanhf(g[2 * D + d]);
    float o_ = sigmoidf_(g[3 * D + d]);
    float cv = f_ * c[t] + i_ * g_;
    c[t] = cv;
    h[t] = o_ * tanhf(cv);
}

__global__ void att_e_kernel(const float* __restrict__ h2, const long long* __restrict__ batch,
                             const float* __restrict__ h, float* __restrict__ ebuf,
                             unsigned* __restrict__ mu) {
    int n = blockIdx.x * blockDim.x + threadIdx.x;
    if (n >= N_NODES) return;
    int g = (int)batch[n];
    const float* xr = h2 + (size_t)n * D;
    const float* q  = h + (size_t)g * D;
    float e = 0.0f;
    for (int k = 0; k < D; ++k) e += xr[k] * q[k];
    ebuf[n] = e;
    atomicMax(&mu[g], fmap_(e));
}

__global__ void att_ex_kernel(float* __restrict__ ebuf, const long long* __restrict__ batch,
                              const unsigned* __restrict__ mu, float* __restrict__ ssum) {
    int n = blockIdx.x * blockDim.x + threadIdx.x;
    if (n >= N_NODES) return;
    int g = (int)batch[n];
    unsigned u = mu[g];
    float m = (u == 0u) ? 0.0f : funmap_(u);           // empty-graph guard
    if (!(m > -3.0e38f && m < 3.0e38f)) m = 0.0f;      // isfinite guard
    float ex = __expf(ebuf[n] - m);
    ebuf[n] = ex;
    atomicAdd(&ssum[g], ex);
}

__global__ void att_r_kernel(const float* __restrict__ ebuf, const long long* __restrict__ batch,
                             const float* __restrict__ ssum, const float* __restrict__ h2,
                             float* __restrict__ r) {
    int t = blockIdx.x * blockDim.x + threadIdx.x;
    int n = t >> 5;
    if (n >= N_NODES) return;
    int lane = t & 31;
    int g = (int)batch[n];
    float a = ebuf[n] / (ssum[g] + 1e-16f);
    const float4* xr = (const float4*)(h2 + (size_t)n * D);
    float4 v = xr[lane];
    float* rp = r + (size_t)g * D + lane * 4;
    atomicAdd(rp + 0, a * v.x);
    atomicAdd(rp + 1, a * v.y);
    atomicAdd(rp + 2, a * v.z);
    atomicAdd(rp + 3, a * v.w);
}

__global__ void qstar_kernel(const float* __restrict__ h, const float* __restrict__ r,
                             float* __restrict__ qstar) {
    int t = blockIdx.x * blockDim.x + threadIdx.x;
    if (t >= N_GRAPHS * D) return;
    int b = t >> 7, d = t & 127;
    qstar[(size_t)b * 2 * D + d]     = h[t];
    qstar[(size_t)b * 2 * D + D + d] = r[t];
}

// ---------------- orchestration ----------------
static inline int nblk(long long n) { return (int)((n + 255) / 256); }

extern "C" void kernel_launch(void* const* d_in, const int* in_sizes, int n_in,
                              void* d_out, int out_size, void* d_ws, size_t ws_size,
                              hipStream_t stream) {
    const float*     x     = (const float*)d_in[0];
    const long long* ei    = (const long long*)d_in[1];
    const long long* batch = (const long long*)d_in[2];
    const float*     W1    = (const float*)d_in[3];
    const float*     b1    = (const float*)d_in[4];
    const float*     W2    = (const float*)d_in[5];
    const float*     b2    = (const float*)d_in[6];
    const float*     Wih   = (const float*)d_in[7];
    const float*     Whh   = (const float*)d_in[8];
    const float*     bih   = (const float*)d_in[9];
    const float*     bhh   = (const float*)d_in[10];
    float* out = (float*)d_out;   // q_star lives here: [256, 256]

    float* ws = (float*)d_ws;
    size_t off = 0;
    float* dinv  = ws + off; off += 50048;
    float* bufA  = ws + off; off += (size_t)N_NODES * D;   // xw
    float* bufB  = ws + off; off += (size_t)N_NODES * D;   // agg / h1 / h2
    float* gates = ws + off; off += (size_t)N_GRAPHS * 4 * D;
    float* hS    = ws + off; off += (size_t)N_GRAPHS * D;
    float* cS    = ws + off; off += (size_t)N_GRAPHS * D;
    float* ebuf  = ws + off; off += 50048;
    unsigned* mu = (unsigned*)(ws + off); off += 256;
    float* ssum  = ws + off; off += 256;
    float* rS    = ws + off; off += (size_t)N_GRAPHS * D;

    // degrees: deg = 1 (self loop) + in-degree, then dinv = rsqrt(deg)
    fill_f32<<<nblk(N_NODES), 256, 0, stream>>>(dinv, 1.0f, N_NODES);
    deg_kernel<<<nblk(N_EDGES), 256, 0, stream>>>(ei, dinv);
    rsqrt_kernel<<<nblk(N_NODES), 256, 0, stream>>>(dinv);

    // ---- GCN layer 1: h1 = relu(norm-agg(x@W1) + b1) ----
    gemm_wmma_kernel<<<N_NODES / 16, 256, 0, stream>>>(x, W1, bufA);
    fill_f32<<<nblk((long long)N_NODES * D), 256, 0, stream>>>(bufB, 0.0f, N_NODES * D);
    scatter_kernel<<<nblk((long long)N_EDGES * 32), 256, 0, stream>>>(bufA, ei, dinv, bufB);
    self_bias_kernel<<<nblk((long long)N_NODES * D), 256, 0, stream>>>(bufB, bufA, dinv, b1, 1);

    // ---- GCN layer 2: h2 = norm-agg(h1@W2) + b2 ----
    gemm_wmma_kernel<<<N_NODES / 16, 256, 0, stream>>>(bufB, W2, bufA);
    fill_f32<<<nblk((long long)N_NODES * D), 256, 0, stream>>>(bufB, 0.0f, N_NODES * D);
    scatter_kernel<<<nblk((long long)N_EDGES * 32), 256, 0, stream>>>(bufA, ei, dinv, bufB);
    self_bias_kernel<<<nblk((long long)N_NODES * D), 256, 0, stream>>>(bufB, bufA, dinv, b2, 0);

    // ---- Set2Set ----
    fill_f32<<<nblk(N_GRAPHS * D), 256, 0, stream>>>(hS, 0.0f, N_GRAPHS * D);
    fill_f32<<<nblk(N_GRAPHS * D), 256, 0, stream>>>(cS, 0.0f, N_GRAPHS * D);
    fill_f32<<<nblk(N_GRAPHS * 2 * D), 256, 0, stream>>>(out, 0.0f, N_GRAPHS * 2 * D);

    for (int step = 0; step < STEPS; ++step) {
        gates_kernel<<<nblk(N_GRAPHS * 4 * D), 256, 0, stream>>>(out, hS, Wih, Whh, bih, bhh, gates);
        lstm_kernel<<<nblk(N_GRAPHS * D), 256, 0, stream>>>(gates, hS, cS);

        fill_f32<<<1, 256, 0, stream>>>((float*)mu, 0.0f, N_GRAPHS);   // encoded -inf == 0
        fill_f32<<<1, 256, 0, stream>>>(ssum, 0.0f, N_GRAPHS);
        fill_f32<<<nblk(N_GRAPHS * D), 256, 0, stream>>>(rS, 0.0f, N_GRAPHS * D);

        att_e_kernel<<<nblk(N_NODES), 256, 0, stream>>>(bufB, batch, hS, ebuf, mu);
        att_ex_kernel<<<nblk(N_NODES), 256, 0, stream>>>(ebuf, batch, mu, ssum);
        att_r_kernel<<<nblk((long long)N_NODES * 32), 256, 0, stream>>>(ebuf, batch, ssum, bufB, rS);
        qstar_kernel<<<nblk(N_GRAPHS * D), 256, 0, stream>>>(hS, rS, out);
    }
}
